// ComplexCNNLSTM_35811437314268
// MI455X (gfx1250) — matrix operand order repaired and checked by reading, not compiled, not run on hardware
//
#include <hip/hip_runtime.h>

typedef __attribute__((ext_vector_type(16))) __bf16 v16bf;
typedef __attribute__((ext_vector_type(8)))  float  v8f;

// ---------------------------------------------------------------------------
// WMMA helpers (CDNA5 gfx1250, wave32)
// ---------------------------------------------------------------------------
static __device__ __forceinline__ v8f wmma_bf16(v16bf a, v16bf b, v8f c) {
  // emits v_wmma_f32_16x16x32_bf16
  return __builtin_amdgcn_wmma_f32_16x16x32_bf16(false, a, false, b, (short)0, c,
                                                 false, false);
}

// A fragment: 16x32 bf16, row-major A[m, k], lda elements per row.
// Layout (ISA 7.12.2): lane m=lane&15; VGPR0-3 hold K=kb..kb+7, VGPR4-7 K=kb+16..kb+23,
// where kb = (lane>>4)*8.  Works for global or LDS pointers.
static __device__ __forceinline__ v16bf load_a_frag(const __bf16* A, int lda,
                                                    int m0, int k0, int lane) {
  int m  = m0 + (lane & 15);
  int kb = k0 + ((lane >> 4) << 3);
  const __bf16* p = A + (size_t)m * lda + kb;
  union { v16bf v; uint4 q[2]; } u;
  u.q[0] = *reinterpret_cast<const uint4*>(p);
  u.q[1] = *reinterpret_cast<const uint4*>(p + 16);
  return u.v;
}

// B fragment: 32x16 bf16 of matrix B[k, n], where memory holds W[n, k] row-major
// (ldw elements per row), i.e. B = W^T. Lane n=lane&15 holds K=kb..kb+7 and
// kb+16..kb+23, where kb = (lane>>4)*8 (same order as the A fragment).
static __device__ __forceinline__ v16bf load_bw_frag(const __bf16* W, int ldw,
                                                     int k0, int n0, int lane) {
  int n  = n0 + (lane & 15);
  int kb = k0 + ((lane >> 4) << 3);
  const __bf16* p = W + (size_t)n * ldw + kb;
  union { v16bf v; uint4 q[2]; } u;
  u.q[0] = *reinterpret_cast<const uint4*>(p);
  u.q[1] = *reinterpret_cast<const uint4*>(p + 16);
  return u.v;
}

// ---------------------------------------------------------------------------
// CDNA5 async global->LDS copy (ASYNCcnt path, ISA 08_async_tensor)
// ---------------------------------------------------------------------------
static __device__ __forceinline__ void async_copy_b128(unsigned lds_addr,
                                                       const void* gaddr) {
  asm volatile("global_load_async_to_lds_b128 %0, %1, off"
               :: "v"(lds_addr), "v"(gaddr) : "memory");
}
static __device__ __forceinline__ void wait_async0() {
  asm volatile("s_wait_asynccnt 0x0" ::: "memory");
}

// ---------------------------------------------------------------------------
// Weight packing
// ---------------------------------------------------------------------------
// conv weights [256,256,3] f32 -> A matrices [256, 768] bf16 with kappa = k*256+ci
__global__ void pack_conv_w_kernel(const float* __restrict__ wr,
                                   const float* __restrict__ wi,
                                   __bf16* Ar, __bf16* Ai, __bf16* Ain) {
  int idx = blockIdx.x * blockDim.x + threadIdx.x;
  if (idx >= 256 * 768) return;
  int co = idx / 768, rem = idx % 768;
  int k = rem >> 8, ci = rem & 255;
  float vr = wr[(co * 256 + ci) * 3 + k];
  float vi = wi[(co * 256 + ci) * 3 + k];
  Ar[idx]  = (__bf16)vr;
  Ai[idx]  = (__bf16)vi;
  Ain[idx] = (__bf16)(-vi);
}

__global__ void pack_mat_kernel(const float* __restrict__ w, __bf16* dst,
                                __bf16* dneg, int n) {
  int i = blockIdx.x * blockDim.x + threadIdx.x;
  if (i >= n) return;
  float v = w[i];
  dst[i] = (__bf16)v;
  if (dneg) dneg[i] = (__bf16)(-v);
}

__global__ void f32_to_bf16_kernel(const float* __restrict__ src, __bf16* dst, int n) {
  int i = blockIdx.x * blockDim.x + threadIdx.x;
  if (i < n) dst[i] = (__bf16)src[i];
}

// ---------------------------------------------------------------------------
// Input complex conv: C=8 -> H=256, K=3, SAME. x is [B, T, 8]. out [B,256,1024].
// ---------------------------------------------------------------------------
__global__ __launch_bounds__(256) void conv_in_kernel(
    const float* __restrict__ xr, const float* __restrict__ xi,
    const float* __restrict__ wr, const float* __restrict__ wi,
    const float* __restrict__ br, const float* __restrict__ bi,
    float* __restrict__ yr, float* __restrict__ yi) {
  int idx = blockIdx.x * blockDim.x + threadIdx.x;  // [B,H,T] linear
  int t = idx & 1023;
  int h = (idx >> 10) & 255;
  int b = idx >> 18;
  float sr = 0.f, si = 0.f;
#pragma unroll
  for (int k = 0; k < 3; ++k) {
    int tp = t + k - 1;
    if ((unsigned)tp < 1024u) {
      const float* xrp = xr + ((size_t)b * 1024 + tp) * 8;
      const float* xip = xi + ((size_t)b * 1024 + tp) * 8;
#pragma unroll
      for (int c = 0; c < 8; ++c) {
        float w_r = wr[(h * 8 + c) * 3 + k];
        float w_i = wi[(h * 8 + c) * 3 + k];
        float a_r = xrp[c], a_i = xip[c];
        sr += w_r * a_r - w_i * a_i;
        si += w_r * a_i + w_i * a_r;
      }
    }
  }
  yr[idx] = sr + (br[h] - bi[h]);
  yi[idx] = si + (br[h] + bi[h]);
}

// ---------------------------------------------------------------------------
// GroupNorm (G=8 over [B,256,1024] f32) + ReLU + optional residual.
// f32 out keeps [B,H,T]; bf16 out is written TRANSPOSED to [B,T,H] so the
// conv GEMM / RNN GEMMs read channel-contiguous rows.
// ---------------------------------------------------------------------------
__global__ __launch_bounds__(256) void gnorm_relu_kernel(
    const float* __restrict__ in, const float* __restrict__ gam,
    const float* __restrict__ bet, const float* __restrict__ res,
    float* outf, __bf16* outbf) {
  int bg = blockIdx.x;
  int b = bg >> 3, g = bg & 7;
  int tid = threadIdx.x;
  __shared__ float s1[256], s2[256];
  size_t base = ((size_t)b * 256 + g * 32) * 1024;
  float sum = 0.f, sq = 0.f;
  for (int j = tid; j < 32768; j += 256) {
    float v = in[base + j];
    sum += v;
    sq += v * v;
  }
  s1[tid] = sum;
  s2[tid] = sq;
  __syncthreads();
  for (int s = 128; s > 0; s >>= 1) {
    if (tid < s) { s1[tid] += s1[tid + s]; s2[tid] += s2[tid + s]; }
    __syncthreads();
  }
  float mu  = s1[0] * (1.f / 32768.f);
  float var = s2[0] * (1.f / 32768.f) - mu * mu;
  float rs  = rsqrtf(var + 1e-5f);
  for (int j = tid; j < 32768; j += 256) {
    int cl = j >> 10;             // channel within group
    int t  = j & 1023;
    int c  = (g << 5) + cl;
    size_t off = base + j;        // [B,H,T]
    float v = (in[off] - mu) * rs * gam[c] + bet[c];
    v = fmaxf(v, 0.f);
    if (res) v += res[off];
    if (outf) outf[off] = v;
    if (outbf) outbf[((size_t)b * 1024 + t) * 256 + c] = (__bf16)v;  // [B,T,H]
  }
}

// ---------------------------------------------------------------------------
// Complex conv H->H as implicit GEMM with WMMA bf16 + async LDS staging.
//   Y^T[t, co] = X^T[t, kappa] x W^T[kappa, co], kappa = k*256 + ci,
//   conv taps are t-row shifts of the staged tile (halo = 1 row each side).
//   y_r = Xr*Wr + Xi*(-Wi) ; y_i = Xi*Wr + Xr*Wi.
// Block = 128 threads (4 waves), tile 128t x 32co. Per 32-channel chunk the
// activation slab [130 x 32] (t-major, channel-contiguous) is staged with
// GLOBAL_LOAD_ASYNC_TO_LDS_B128 and reused by all 4 waves and all 3 taps;
// A-fragments are two aligned ds_load_b128 per lane (pitch 40 -> no bank dup).
// ---------------------------------------------------------------------------
#define XPITCH 40

__global__ __launch_bounds__(128) void cconv_gemm_kernel(
    const __bf16* __restrict__ xr, const __bf16* __restrict__ xi,  // [B,1024,256]
    const __bf16* __restrict__ Ar, const __bf16* __restrict__ Ai,
    const __bf16* __restrict__ Ain,                                // [256,768]
    const float* __restrict__ br, const float* __restrict__ bi,
    float* __restrict__ yr, float* __restrict__ yi) {              // [B,256,1024]
  __shared__ __align__(16) __bf16 sXr[130 * XPITCH];
  __shared__ __align__(16) __bf16 sXi[130 * XPITCH];
  int tid  = threadIdx.x;
  int lane = tid & 31;
  int w    = tid >> 5;            // wave 0..3 -> t-slice
  int t0   = blockIdx.x * 128;
  int co0  = blockIdx.y * 32;
  int b    = blockIdx.z;
  unsigned ldsR = (unsigned)(uintptr_t)&sXr[0];
  unsigned ldsI = (unsigned)(uintptr_t)&sXi[0];

  v8f accr[2][2] = {};
  v8f acci[2][2] = {};

  for (int cc = 0; cc < 8; ++cc) {
    int c0 = cc * 32;
    __syncthreads();  // previous chunk fully consumed by all waves
    // stage 130 t-rows x 32 channels (4 b128 segs/row), rows t0-1 .. t0+128
    for (int s = tid; s < 520; s += 128) {
      int row = s >> 2, seg = s & 3;
      ptrdiff_t grow = (ptrdiff_t)b * 1024 + (t0 - 1 + row);
      ptrdiff_t goff = grow * 256 + c0 + seg * 8;
      unsigned loff = (unsigned)((row * XPITCH + seg * 8) * 2);
      async_copy_b128(ldsR + loff, (const void*)(xr + goff));
      async_copy_b128(ldsI + loff, (const void*)(xi + goff));
    }
    wait_async0();
    __syncthreads();  // all waves' async data landed in LDS
    if (t0 == 0 && tid < 32) {        // conv zero-padding rows
      sXr[tid] = (__bf16)0.0f;
      sXi[tid] = (__bf16)0.0f;
    }
    if (t0 == 896 && tid < 32) {
      sXr[129 * XPITCH + tid] = (__bf16)0.0f;
      sXi[129 * XPITCH + tid] = (__bf16)0.0f;
    }
    __syncthreads();
#pragma unroll
    for (int k = 0; k < 3; ++k) {
      int kap0 = k * 256 + c0;
      if (k == 0)
        __builtin_prefetch(Ar + (size_t)co0 * 768 + ((cc + 1) & 7) * 32, 0, 1);
      v16bf fXr[2], fXi[2];
#pragma unroll
      for (int m = 0; m < 2; ++m) {
        int r0 = w * 32 + m * 16 + k;  // local t row base incl. tap shift
        fXr[m] = load_a_frag(sXr, XPITCH, r0, 0, lane);
        fXi[m] = load_a_frag(sXi, XPITCH, r0, 0, lane);
      }
      v16bf fWr[2], fWi[2], fWin[2];
#pragma unroll
      for (int n = 0; n < 2; ++n) {
        fWr[n]  = load_bw_frag(Ar,  768, kap0, co0 + n * 16, lane);
        fWi[n]  = load_bw_frag(Ai,  768, kap0, co0 + n * 16, lane);
        fWin[n] = load_bw_frag(Ain, 768, kap0, co0 + n * 16, lane);
      }
#pragma unroll
      for (int m = 0; m < 2; ++m)
#pragma unroll
        for (int n = 0; n < 2; ++n) {
          accr[m][n] = wmma_bf16(fXr[m], fWr[n],  accr[m][n]);
          accr[m][n] = wmma_bf16(fXi[m], fWin[n], accr[m][n]);
          acci[m][n] = wmma_bf16(fXi[m], fWr[n],  acci[m][n]);
          acci[m][n] = wmma_bf16(fXr[m], fWi[n],  acci[m][n]);
        }
    }
  }
  // D = Y^T tile: rows = t, cols = co ; store back to [B,256,1024] f32
#pragma unroll
  for (int m = 0; m < 2; ++m) {
    int tb = t0 + w * 32 + m * 16 + ((lane >> 4) << 3);
#pragma unroll
    for (int n = 0; n < 2; ++n) {
      int co = co0 + n * 16 + (lane & 15);
      float biasr = br[co] - bi[co];
      float biasi = br[co] + bi[co];
#pragma unroll
      for (int e = 0; e < 8; ++e) {
        size_t off = ((size_t)b * 256 + co) * 1024 + (tb + e);
        yr[off] = accr[m][n][e] + biasr;
        yi[off] = acci[m][n][e] + biasi;
      }
    }
  }
}

// ---------------------------------------------------------------------------
// Transpose [B,256,1024] -> [B,1024,256] (f32 sequence for the RNN phase)
// ---------------------------------------------------------------------------
__global__ void transpose_kernel(const float* __restrict__ ir,
                                 const float* __restrict__ ii,
                                 float* __restrict__ outr, float* __restrict__ outi) {
  __shared__ float tr[32][33];
  __shared__ float ti[32][33];
  int b  = blockIdx.z;
  int h0 = blockIdx.y * 32;
  int t0 = blockIdx.x * 32;
  int tx = threadIdx.x, ty = threadIdx.y;
  for (int j = ty; j < 32; j += 8) {
    size_t off = ((size_t)b * 256 + h0 + j) * 1024 + t0 + tx;
    tr[j][tx] = ir[off];
    ti[j][tx] = ii[off];
  }
  __syncthreads();
  for (int j = ty; j < 32; j += 8) {
    size_t off = ((size_t)b * 1024 + t0 + j) * 256 + h0 + tx;
    outr[off] = tr[tx][j];
    outi[off] = ti[tx][j];
  }
}

// ---------------------------------------------------------------------------
// RNN input projections (batched over B*T):
//   P_r = r@Wir^T + i@(-Wii)^T + (b_ir - b_ii)
//   P_i = i@Wir^T + r@Wii^T    + (b_ir + b_ii)
// ---------------------------------------------------------------------------
__global__ __launch_bounds__(32) void rnn_p_gemm_kernel(
    const __bf16* __restrict__ sr, const __bf16* __restrict__ si,
    const __bf16* __restrict__ Wir, const __bf16* __restrict__ Wii,
    const __bf16* __restrict__ Wiin,
    const float* __restrict__ b_ir, const float* __restrict__ b_ii,
    float* __restrict__ Pr, float* __restrict__ Pi) {
  int lane = threadIdx.x;
  int mu0 = blockIdx.x * 32;
  int n0  = blockIdx.y * 32;
  v8f pr[2][2] = {};
  v8f pi[2][2] = {};
  for (int ks = 0; ks < 8; ++ks) {
    int k0 = ks * 32;
    v16bf fr[2], fi[2];
#pragma unroll
    for (int m = 0; m < 2; ++m) {
      fr[m] = load_a_frag(sr, 256, mu0 + m * 16, k0, lane);
      fi[m] = load_a_frag(si, 256, mu0 + m * 16, k0, lane);
    }
    v16bf bir[2], bii[2], biin[2];
#pragma unroll
    for (int n = 0; n < 2; ++n) {
      bir[n]  = load_bw_frag(Wir,  256, k0, n0 + n * 16, lane);
      bii[n]  = load_bw_frag(Wii,  256, k0, n0 + n * 16, lane);
      biin[n] = load_bw_frag(Wiin, 256, k0, n0 + n * 16, lane);
    }
#pragma unroll
    for (int m = 0; m < 2; ++m)
#pragma unroll
      for (int n = 0; n < 2; ++n) {
        pr[m][n] = wmma_bf16(fr[m], bir[n],  pr[m][n]);
        pr[m][n] = wmma_bf16(fi[m], biin[n], pr[m][n]);
        pi[m][n] = wmma_bf16(fi[m], bir[n],  pi[m][n]);
        pi[m][n] = wmma_bf16(fr[m], bii[n],  pi[m][n]);
      }
  }
#pragma unroll
  for (int m = 0; m < 2; ++m) {
    int mb = mu0 + m * 16 + ((lane >> 4) << 3);
#pragma unroll
    for (int n = 0; n < 2; ++n) {
      int nn = n0 + n * 16 + (lane & 15);
      float biasr = b_ir[nn] - b_ii[nn];
      float biasi = b_ir[nn] + b_ii[nn];
#pragma unroll
      for (int e = 0; e < 8; ++e) {
        size_t off = (size_t)(mb + e) * 256 + nn;
        Pr[off] = pr[m][n][e] + biasr;
        Pi[off] = pi[m][n][e] + biasi;
      }
    }
  }
}

// ---------------------------------------------------------------------------
// Sequential Elman recurrence: one workgroup (32 waves), h in LDS (bf16).
//   r_t = tanh(P_r + h_r@Whr^T + h_i@(-Whi)^T + b_hr - b_hi)
//   i_t = tanh(P_i + h_i@Whr^T + h_r@Whi^T    + b_hr + b_hi)
// seq is updated in place with residual: seq += tanh-output.
// ---------------------------------------------------------------------------
__global__ __launch_bounds__(1024) void rnn_seq_kernel(
    float* __restrict__ seq_r, float* __restrict__ seq_i,
    const float* __restrict__ P_r, const float* __restrict__ P_i,
    const __bf16* __restrict__ Whr, const __bf16* __restrict__ Whi,
    const __bf16* __restrict__ Whin,
    const float* __restrict__ b_hr, const float* __restrict__ b_hi) {
  __shared__ __align__(16) __bf16 h_r[32 * 256];
  __shared__ __align__(16) __bf16 h_i[32 * 256];
  int tid = threadIdx.x;
  for (int idx = tid; idx < 32 * 256; idx += 1024) {
    h_r[idx] = (__bf16)0.0f;
    h_i[idx] = (__bf16)0.0f;
  }
  __syncthreads();

  int wave = tid >> 5, lane = tid & 31;
  int m0 = (wave & 1) * 16;       // batch tile
  int n0 = (wave >> 1) * 16;      // hidden tile
  int n  = n0 + (lane & 15);
  int mb = m0 + ((lane >> 4) << 3);
  float hbias_r = b_hr[n] - b_hi[n];
  float hbias_i = b_hr[n] + b_hi[n];

  for (int t = 0; t < 1024; ++t) {
    v8f ar = {};
    v8f ai = {};
#pragma unroll
    for (int ks = 0; ks < 8; ++ks) {
      int k0 = ks * 32;
      v16bf fhr = load_a_frag(h_r, 256, m0, k0, lane);
      v16bf fhi = load_a_frag(h_i, 256, m0, k0, lane);
      v16bf bhr = load_bw_frag(Whr,  256, k0, n0, lane);
      v16bf bhi = load_bw_frag(Whi,  256, k0, n0, lane);
      v16bf bhn = load_bw_frag(Whin, 256, k0, n0, lane);
      ar = wmma_bf16(fhr, bhr, ar);
      ar = wmma_bf16(fhi, bhn, ar);
      ai = wmma_bf16(fhi, bhr, ai);
      ai = wmma_bf16(fhr, bhi, ai);
    }
    __syncthreads();  // all waves finished reading h for this step
#pragma unroll
    for (int e = 0; e < 8; ++e) {
      int b = mb + e;
      size_t off = ((size_t)b * 1024 + t) * 256 + n;
      float vr = tanhf(ar[e] + P_r[off] + hbias_r);
      float vi = tanhf(ai[e] + P_i[off] + hbias_i);
      h_r[b * 256 + n] = (__bf16)vr;
      h_i[b * 256 + n] = (__bf16)vi;
      seq_r[off] += vr;  // residual: tanh-output + input sequence
      seq_i[off] += vi;
    }
    __syncthreads();  // h updated before next step reads it
  }
}

// ---------------------------------------------------------------------------
// Final LayerNorm over H=256 at t = T-1
// ---------------------------------------------------------------------------
__global__ __launch_bounds__(256) void lnorm_kernel(const float* __restrict__ seq,
                                                    const float* __restrict__ g,
                                                    const float* __restrict__ bta,
                                                    float* __restrict__ out) {
  int b = blockIdx.x, tid = threadIdx.x;
  float x = seq[((size_t)b * 1024 + 1023) * 256 + tid];
  __shared__ float s1[256], s2[256];
  s1[tid] = x;
  s2[tid] = x * x;
  __syncthreads();
  for (int s = 128; s > 0; s >>= 1) {
    if (tid < s) { s1[tid] += s1[tid + s]; s2[tid] += s2[tid + s]; }
    __syncthreads();
  }
  float mu  = s1[0] * (1.f / 256.f);
  float var = s2[0] * (1.f / 256.f) - mu * mu;
  out[b * 256 + tid] = (x - mu) * rsqrtf(var + 1e-5f) * g[tid] + bta[tid];
}

// ---------------------------------------------------------------------------
// Host launcher
// ---------------------------------------------------------------------------
extern "C" void kernel_launch(void* const* d_in, const int* in_sizes, int n_in,
                              void* d_out, int out_size, void* d_ws, size_t ws_size,
                              hipStream_t stream) {
  (void)in_sizes; (void)out_size; (void)ws_size;
  auto F = [&](int i) -> const float* {
    int j = (i < n_in) ? i : (n_in > 0 ? n_in - 1 : 0);
    return (const float*)d_in[j];
  };
  char* ws = (char*)d_ws;
  const size_t A4 = 33554432ull;  // [B,256,1024] f32 bytes
  const size_t A2 = 16777216ull;  // [B,256,1024] bf16 bytes
  float*  cur_r = (float*)(ws);
  float*  cur_i = (float*)(ws + A4);
  float*  cnv_r = (float*)(ws + 2 * A4);   // also seq_r after transpose
  float*  cnv_i = (float*)(ws + 3 * A4);   // also seq_i
  __bf16* bf_r  = (__bf16*)(ws + 4 * A4);  // bf16 activations, [B,T,H] layout
  __bf16* bf_i  = (__bf16*)(ws + 4 * A4 + A2);
  char* wb = ws + 4 * A4 + 2 * A2;
  const size_t CW = 256 * 768 * 2;  // conv A matrix bytes
  const size_t RW = 256 * 256 * 2;  // rnn W matrix bytes
  __bf16 *cAr[4], *cAi[4], *cAin[4];
  for (int i = 0; i < 4; ++i) {
    cAr[i]  = (__bf16*)(wb + i * 3 * CW);
    cAi[i]  = (__bf16*)(wb + i * 3 * CW + CW);
    cAin[i] = (__bf16*)(wb + i * 3 * CW + 2 * CW);
  }
  char* rb = wb + 12 * CW;
  __bf16* rW[2][6];  // Wir, Wii, Wiin, Whr, Whi, Whin per layer
  for (int l = 0; l < 2; ++l)
    for (int j = 0; j < 6; ++j)
      rW[l][j] = (__bf16*)(rb + (size_t)(l * 6 + j) * RW);

  // Input index map (jax pytree flatten, sorted dict keys):
  // block1: [0..15] conv1_i.b/w, conv1_r.b/w, conv2_i.b/w, conv2_r.b/w,
  //         gn_i1.b/g, gn_i2.b/g, gn_r1.b/g, gn_r2.b/g ; block2: [16..31]
  // 32..35 conv_in_i.b/w, conv_in_r.b/w ; 36..39 gn_in_i.b/g, gn_in_r.b/g
  // 40..43 ln_i.b/g, ln_r.b/g ; 44..51 rnn0 (W_hi.b/w, W_hr.b/w, W_ii.b/w, W_ir.b/w)
  // 52..59 rnn1 ; 60 x_imag ; 61 x_real

  // ---- pack weights to bf16 (WMMA layouts) ----
  const int cwidx[4][2] = {{3, 1}, {7, 5}, {19, 17}, {23, 21}};  // (wr, wi)
  for (int i = 0; i < 4; ++i)
    pack_conv_w_kernel<<<768, 256, 0, stream>>>(F(cwidx[i][0]), F(cwidx[i][1]),
                                                cAr[i], cAi[i], cAin[i]);
  for (int l = 0; l < 2; ++l) {
    int base = 44 + l * 8;
    pack_mat_kernel<<<256, 256, 0, stream>>>(F(base + 7), rW[l][0], (__bf16*)nullptr, 65536);
    pack_mat_kernel<<<256, 256, 0, stream>>>(F(base + 5), rW[l][1], rW[l][2], 65536);
    pack_mat_kernel<<<256, 256, 0, stream>>>(F(base + 3), rW[l][3], (__bf16*)nullptr, 65536);
    pack_mat_kernel<<<256, 256, 0, stream>>>(F(base + 1), rW[l][4], rW[l][5], 65536);
  }

  // ---- input conv + gnorm ----
  conv_in_kernel<<<32768, 256, 0, stream>>>(F(61), F(60), F(35), F(33), F(34), F(32),
                                            cnv_r, cnv_i);
  gnorm_relu_kernel<<<256, 256, 0, stream>>>(cnv_r, F(39), F(38), nullptr, cur_r, bf_r);
  gnorm_relu_kernel<<<256, 256, 0, stream>>>(cnv_i, F(37), F(36), nullptr, cur_i, bf_i);

  // ---- two residual complex conv blocks ----
  for (int blk = 0; blk < 2; ++blk) {
    int base = blk * 16;
    int c1 = blk * 2, c2 = blk * 2 + 1;
    cconv_gemm_kernel<<<dim3(8, 8, 32), 128, 0, stream>>>(
        bf_r, bf_i, cAr[c1], cAi[c1], cAin[c1], F(base + 2), F(base + 0), cnv_r, cnv_i);
    gnorm_relu_kernel<<<256, 256, 0, stream>>>(cnv_r, F(base + 13), F(base + 12),
                                               nullptr, (float*)nullptr, bf_r);
    gnorm_relu_kernel<<<256, 256, 0, stream>>>(cnv_i, F(base + 9), F(base + 8),
                                               nullptr, (float*)nullptr, bf_i);
    cconv_gemm_kernel<<<dim3(8, 8, 32), 128, 0, stream>>>(
        bf_r, bf_i, cAr[c2], cAi[c2], cAin[c2], F(base + 6), F(base + 4), cnv_r, cnv_i);
    gnorm_relu_kernel<<<256, 256, 0, stream>>>(cnv_r, F(base + 15), F(base + 14),
                                               cur_r, cur_r, bf_r);
    gnorm_relu_kernel<<<256, 256, 0, stream>>>(cnv_i, F(base + 11), F(base + 10),
                                               cur_i, cur_i, bf_i);
  }

  // ---- transpose f32 sequence to [B,T,H] (into cnv buffers => seq) ----
  // bf_r/bf_i already hold the bf16 [B,T,H] sequence (written by last gnorm).
  transpose_kernel<<<dim3(32, 8, 32), dim3(32, 8), 0, stream>>>(cur_r, cur_i, cnv_r, cnv_i);

  // ---- RNN layers ----
  for (int l = 0; l < 2; ++l) {
    int base = 44 + l * 8;
    if (l > 0) {  // layer 0's bf16 seq was produced by the final gnorm
      f32_to_bf16_kernel<<<32768, 256, 0, stream>>>(cnv_r, bf_r, 8388608);
      f32_to_bf16_kernel<<<32768, 256, 0, stream>>>(cnv_i, bf_i, 8388608);
    }
    rnn_p_gemm_kernel<<<dim3(1024, 8), 32, 0, stream>>>(
        bf_r, bf_i, rW[l][0], rW[l][1], rW[l][2], F(base + 6), F(base + 4),
        cur_r, cur_i);
    rnn_seq_kernel<<<1, 1024, 0, stream>>>(cnv_r, cnv_i, cur_r, cur_i,
                                           rW[l][3], rW[l][4], rW[l][5],
                                           F(base + 2), F(base + 0));
  }

  // ---- final LayerNorm on last timestep ----
  float* out = (float*)d_out;
  lnorm_kernel<<<32, 256, 0, stream>>>(cnv_r, F(43), F(42), out);
  lnorm_kernel<<<32, 256, 0, stream>>>(cnv_i, F(41), F(40), out + 8192);
}
